// ResGINE_85237920956548
// MI455X (gfx1250) — compile-verified, matrix operand order checked by reading, not run on hardware
//
#include <hip/hip_runtime.h>
#include <hip/hip_bf16.h>
#include <math.h>
#include <stdint.h>

typedef __attribute__((ext_vector_type(16))) _Float16 v16h;
typedef __attribute__((ext_vector_type(8)))  float    v8f;
typedef __attribute__((ext_vector_type(4)))  int      v4i;

#define N_BLOCKS 15
#define HID 128
#define EDGE_DIM 8
#define MT_PER_BLK 2

// ---------------------------------------------------------------------------
// CDNA5 async global->LDS copy helpers (guarded; fall back to sync copy)
// builtin signature: (v4i addrspace(1)*, v4i addrspace(3)*, imm offset, imm cpol)
// ---------------------------------------------------------------------------
#if __has_builtin(__builtin_amdgcn_global_load_async_to_lds_b128)
#define HAVE_ASYNC_LDS 1
__device__ __forceinline__ void async_copy16(const float* g, float* l) {
    __builtin_amdgcn_global_load_async_to_lds_b128(
        (__attribute__((address_space(1))) v4i*)(g),
        (__attribute__((address_space(3))) v4i*)(l), 0, 0);
}
#else
#define HAVE_ASYNC_LDS 0
__device__ __forceinline__ void async_copy16(const float* g, float* l) {
    *(float4*)l = *(const float4*)g;
}
#endif

#if __has_builtin(__builtin_amdgcn_s_wait_asynccnt)
#define ASYNC_WAIT(n) __builtin_amdgcn_s_wait_asynccnt((unsigned short)(n))
#else
#define ASYNC_WAIT(n) asm volatile("s_wait_asynccnt %0" ::"i"(n) : "memory")
#endif

// ---------------------------------------------------------------------------
// elementwise log1p
// ---------------------------------------------------------------------------
__global__ void k_log1p(const float* __restrict__ in, float* __restrict__ out, int n) {
    int i = blockIdx.x * blockDim.x + threadIdx.x;
    if (i < n) out[i] = logf(in[i] + 1.0f);
}

// h = xlog @ embed_W + embed_b   ([N,16] x [16,128])
__global__ void k_embed(const float* __restrict__ xlog, const float* __restrict__ W,
                        const float* __restrict__ b, float* __restrict__ h, int n_nodes) {
    int i = blockIdx.x * blockDim.x + threadIdx.x;
    if (i >= n_nodes * HID) return;
    int m = i >> 7, n = i & 127;
    float acc = b[n];
    const float* xr = xlog + m * 16;
#pragma unroll
    for (int k = 0; k < 16; ++k) acc = fmaf(xr[k], W[k * HID + n], acc);
    h[i] = acc;
}

// ---------------------------------------------------------------------------
// CSR-by-destination construction (once per call)
// ---------------------------------------------------------------------------
__global__ void k_count(const int* __restrict__ dst, int* __restrict__ counts, int n_edges) {
    int e = blockIdx.x * blockDim.x + threadIdx.x;
    if (e < n_edges) atomicAdd(&counts[dst[e]], 1);
}

__global__ void k_scan(const int* __restrict__ counts, int* __restrict__ row_start,
                       int* __restrict__ cursor, int n) {
    __shared__ int sdata[1024];
    __shared__ int carry;
    if (threadIdx.x == 0) carry = 0;
    __syncthreads();
    for (int base = 0; base < n; base += 1024) {
        int i = base + (int)threadIdx.x;
        int v = (i < n) ? counts[i] : 0;
        sdata[threadIdx.x] = v;
        __syncthreads();
        for (int off = 1; off < 1024; off <<= 1) {
            int t = (threadIdx.x >= (unsigned)off) ? sdata[threadIdx.x - off] : 0;
            __syncthreads();
            sdata[threadIdx.x] += t;
            __syncthreads();
        }
        int excl = sdata[threadIdx.x] - v + carry;
        if (i < n) { row_start[i] = excl; cursor[i] = excl; }
        __syncthreads();
        if (threadIdx.x == 1023) carry += sdata[1023];
        __syncthreads();
    }
    if (threadIdx.x == 0) row_start[n] = carry;
}

__global__ void k_fill(const int* __restrict__ dst, int* __restrict__ cursor,
                       int* __restrict__ edge_ids, int n_edges) {
    int e = blockIdx.x * blockDim.x + threadIdx.x;
    if (e >= n_edges) return;
    int p = atomicAdd(&cursor[dst[e]], 1);
    edge_ids[p] = e;
}

// ---------------------------------------------------------------------------
// wave32 row reductions
// ---------------------------------------------------------------------------
__device__ __forceinline__ float wave_sum(float v) {
#pragma unroll
    for (int off = 16; off > 0; off >>= 1) v += __shfl_xor(v, off, 32);
    return v;
}

__global__ void k_norm(const float* __restrict__ h, float* __restrict__ x2, int n_rows) {
    int wave = (int)((blockIdx.x * blockDim.x + threadIdx.x) >> 5);
    int lane = threadIdx.x & 31;
    if (wave >= n_rows) return;
    float4 v = *(const float4*)(h + (size_t)wave * HID + lane * 4);
    float s = wave_sum(v.x + v.y + v.z + v.w);
    float mean = s * (1.0f / 128.0f);
    float dx = v.x - mean, dy = v.y - mean, dz = v.z - mean, dw = v.w - mean;
    float sq = wave_sum(dx * dx + dy * dy + dz * dz + dw * dw);
    float inv = rsqrtf(sq * (1.0f / 127.0f));
    float4 o = make_float4(v.x * inv, v.y * inv, v.z * inv, v.w * inv);
    *(float4*)(x2 + (size_t)wave * HID + lane * 4) = o;
}

__global__ void k_norm_res(const float* __restrict__ t, float* __restrict__ h, int n_rows) {
    int wave = (int)((blockIdx.x * blockDim.x + threadIdx.x) >> 5);
    int lane = threadIdx.x & 31;
    if (wave >= n_rows) return;
    float4 v = *(const float4*)(t + (size_t)wave * HID + lane * 4);
    float s = wave_sum(v.x + v.y + v.z + v.w);
    float mean = s * (1.0f / 128.0f);
    float dx = v.x - mean, dy = v.y - mean, dz = v.z - mean, dw = v.w - mean;
    float sq = wave_sum(dx * dx + dy * dy + dz * dz + dw * dw);
    float inv = rsqrtf(sq * (1.0f / 127.0f));
    float4* hp = (float4*)(h + (size_t)wave * HID + lane * 4);
    float4 hv = *hp;
    hv.x += v.x * inv; hv.y += v.y * inv; hv.z += v.z * inv; hv.w += v.w * inv;
    *hp = hv;
}

// ---------------------------------------------------------------------------
// GINE aggregation: z[i] = x2[i] + sum_{e: dst=i} relu(x2[src] + ea_log[e]@eW + eb)
// one wave32 per node; lane owns 4 channels; edge weights hoisted to registers
// ---------------------------------------------------------------------------
__global__ void k_aggregate(const float* __restrict__ x2, const float* __restrict__ ea,
                            const int* __restrict__ src_idx, const int* __restrict__ row_start,
                            const int* __restrict__ edge_ids, const float* __restrict__ eW,
                            const float* __restrict__ eb, float* __restrict__ z, int n_nodes) {
    int wave = (int)((blockIdx.x * blockDim.x + threadIdx.x) >> 5);
    int lane = threadIdx.x & 31;
    if (wave >= n_nodes) return;
    int c0 = lane * 4;
    float w[EDGE_DIM][4];
#pragma unroll
    for (int k = 0; k < EDGE_DIM; ++k) {
        float4 wv = *(const float4*)(eW + k * HID + c0);
        w[k][0] = wv.x; w[k][1] = wv.y; w[k][2] = wv.z; w[k][3] = wv.w;
    }
    float4 bb = *(const float4*)(eb + c0);
    float4 acc = *(const float4*)(x2 + (size_t)wave * HID + c0);
    int beg = row_start[wave], end = row_start[wave + 1];
    for (int p = beg; p < end; ++p) {
        int e = edge_ids[p];
        int s = src_idx[e];
        const float4* eap = (const float4*)(ea + (size_t)e * EDGE_DIM);
        float4 e0 = eap[0], e1 = eap[1];
        float em0 = bb.x, em1 = bb.y, em2 = bb.z, em3 = bb.w;
        em0 = fmaf(e0.x, w[0][0], em0); em1 = fmaf(e0.x, w[0][1], em1);
        em2 = fmaf(e0.x, w[0][2], em2); em3 = fmaf(e0.x, w[0][3], em3);
        em0 = fmaf(e0.y, w[1][0], em0); em1 = fmaf(e0.y, w[1][1], em1);
        em2 = fmaf(e0.y, w[1][2], em2); em3 = fmaf(e0.y, w[1][3], em3);
        em0 = fmaf(e0.z, w[2][0], em0); em1 = fmaf(e0.z, w[2][1], em1);
        em2 = fmaf(e0.z, w[2][2], em2); em3 = fmaf(e0.z, w[2][3], em3);
        em0 = fmaf(e0.w, w[3][0], em0); em1 = fmaf(e0.w, w[3][1], em1);
        em2 = fmaf(e0.w, w[3][2], em2); em3 = fmaf(e0.w, w[3][3], em3);
        em0 = fmaf(e1.x, w[4][0], em0); em1 = fmaf(e1.x, w[4][1], em1);
        em2 = fmaf(e1.x, w[4][2], em2); em3 = fmaf(e1.x, w[4][3], em3);
        em0 = fmaf(e1.y, w[5][0], em0); em1 = fmaf(e1.y, w[5][1], em1);
        em2 = fmaf(e1.y, w[5][2], em2); em3 = fmaf(e1.y, w[5][3], em3);
        em0 = fmaf(e1.z, w[6][0], em0); em1 = fmaf(e1.z, w[6][1], em1);
        em2 = fmaf(e1.z, w[6][2], em2); em3 = fmaf(e1.z, w[6][3], em3);
        em0 = fmaf(e1.w, w[7][0], em0); em1 = fmaf(e1.w, w[7][1], em1);
        em2 = fmaf(e1.w, w[7][2], em2); em3 = fmaf(e1.w, w[7][3], em3);
        float4 xs = *(const float4*)(x2 + (size_t)s * HID + c0);
        acc.x += fmaxf(xs.x + em0, 0.0f);
        acc.y += fmaxf(xs.y + em1, 0.0f);
        acc.z += fmaxf(xs.z + em2, 0.0f);
        acc.w += fmaxf(xs.w + em3, 0.0f);
    }
    *(float4*)(z + (size_t)wave * HID + c0) = acc;
}

// ---------------------------------------------------------------------------
// pack W (f32 row-major [128][128]) into WMMA f16 B-fragment layout:
// idx = (((kc*8+nt)*32+lane)*16 + i) ; value = W[kc*32 + kmap(i,lane)][nt*16 + (lane&15)]
// ---------------------------------------------------------------------------
__global__ void k_pack(const float* __restrict__ W1, const float* __restrict__ W2,
                       const float* __restrict__ W3, _Float16* __restrict__ Wp, int total) {
    int tid = blockIdx.x * blockDim.x + threadIdx.x;
    if (tid >= total) return;
    int mi = tid >> 14;            // matrix index 0..44
    int r  = tid & 16383;
    int i    = r & 15;
    int lane = (r >> 4) & 31;
    int t    = r >> 9;             // kc*8 + nt
    int kc = t >> 3, nt = t & 7;
    int k = kc * 32 + ((i < 8) ? 0 : 16) + ((lane >> 4) * 8) + (i & 7);
    int n = nt * 16 + (lane & 15);
    int blk = mi / 3, layer = mi % 3;
    const float* Ws = (layer == 0) ? W1 : (layer == 1) ? W2 : W3;
    Wp[tid] = (_Float16)Ws[(size_t)blk * HID * HID + (size_t)k * HID + n];
}

// ---------------------------------------------------------------------------
// D = act(A @ Wp + bias) ; A [M,128] f32, M%32==0 ; WMMA f32_16x16x32_f16
// 256 threads = 8 waves; wave nt keeps B fragments in registers across
// MT_PER_BLK row-tiles; A tiles double-buffered in LDS via async global->LDS
// ---------------------------------------------------------------------------
__global__ void k_gemm128(const float* __restrict__ A, const _Float16* __restrict__ Wp,
                          const float* __restrict__ bias, float* __restrict__ D, int relu) {
    __shared__ float As[2][16 * HID];
    int tid = threadIdx.x;
    int wave = tid >> 5, lane = tid & 31;
    int nt = wave;

    // B fragments: resident in registers for all tiles of this block
    v16h bf[4];
#pragma unroll
    for (int kc = 0; kc < 4; ++kc)
        bf[kc] = *(const v16h*)(Wp + ((size_t)(kc * 8 + nt) * 32 + lane) * 16);

    int tile0 = blockIdx.x * MT_PER_BLK;
    // prefetch first A tile (2 x 16B chunks per thread)
    {
        const float* g = A + (size_t)tile0 * 16 * HID + tid * 4;
        async_copy16(g,        &As[0][tid * 4]);
        async_copy16(g + 1024, &As[0][tid * 4 + 1024]);
    }

    int arow  = lane & 15;
    int khalf = (lane >> 4) * 8;
    int col   = nt * 16 + (lane & 15);
    int rbase = (lane >> 4) * 8;
    float bv = bias[col];

    for (int it = 0; it < MT_PER_BLK; ++it) {
        int tile = tile0 + it;
        if (it + 1 < MT_PER_BLK) {
            const float* g = A + (size_t)(tile + 1) * 16 * HID + tid * 4;
            async_copy16(g,        &As[(it + 1) & 1][tid * 4]);
            async_copy16(g + 1024, &As[(it + 1) & 1][tid * 4 + 1024]);
            ASYNC_WAIT(2);   // current tile's chunks landed; next tile in flight
        } else {
            ASYNC_WAIT(0);
        }
        __syncthreads();     // all waves' chunks of current tile visible

        const float* ap = &As[it & 1][arow * HID];
        v8f acc = {};
#pragma unroll
        for (int kc = 0; kc < 4; ++kc) {
            int kb = kc * 32 + khalf;
            float4 a0 = *(const float4*)(ap + kb);
            float4 a1 = *(const float4*)(ap + kb + 4);
            float4 a2 = *(const float4*)(ap + kb + 16);
            float4 a3 = *(const float4*)(ap + kb + 20);
            v16h af;
            af[0]  = (_Float16)a0.x; af[1]  = (_Float16)a0.y;
            af[2]  = (_Float16)a0.z; af[3]  = (_Float16)a0.w;
            af[4]  = (_Float16)a1.x; af[5]  = (_Float16)a1.y;
            af[6]  = (_Float16)a1.z; af[7]  = (_Float16)a1.w;
            af[8]  = (_Float16)a2.x; af[9]  = (_Float16)a2.y;
            af[10] = (_Float16)a2.z; af[11] = (_Float16)a2.w;
            af[12] = (_Float16)a3.x; af[13] = (_Float16)a3.y;
            af[14] = (_Float16)a3.z; af[15] = (_Float16)a3.w;
            acc = __builtin_amdgcn_wmma_f32_16x16x32_f16(false, af, false, bf[kc],
                                                         (short)0, acc, false, false);
        }

        float* Dp = D + ((size_t)tile * 16 + rbase) * HID + col;
#pragma unroll
        for (int r = 0; r < 8; ++r) {
            float v = acc[r] + bv;
            if (relu) v = fmaxf(v, 0.0f);
            Dp[(size_t)r * HID] = v;
        }
        __syncthreads();     // all waves done reading this buffer before reuse
    }
}

// ---------------------------------------------------------------------------
// output head: out[i] = sum_o (h[i]/sqrt(15) @ out_W + out_b)[o] * 2^o
// ---------------------------------------------------------------------------
__global__ void k_out(const float* __restrict__ h, const float* __restrict__ oW,
                      const float* __restrict__ ob, float* __restrict__ out, int n_nodes) {
    int wave = (int)((blockIdx.x * blockDim.x + threadIdx.x) >> 5);
    int lane = threadIdx.x & 31;
    if (wave >= n_nodes) return;
    const float inv_s = 0.25819888974716110f; // 1/sqrt(15)
    int c0 = lane * 4;
    float4 hv = *(const float4*)(h + (size_t)wave * HID + c0);
    hv.x *= inv_s; hv.y *= inv_s; hv.z *= inv_s; hv.w *= inv_s;
    float acc[10];
#pragma unroll
    for (int o = 0; o < 10; ++o) {
        acc[o] = hv.x * oW[(c0 + 0) * 10 + o] + hv.y * oW[(c0 + 1) * 10 + o] +
                 hv.z * oW[(c0 + 2) * 10 + o] + hv.w * oW[(c0 + 3) * 10 + o];
    }
#pragma unroll
    for (int off = 16; off > 0; off >>= 1)
#pragma unroll
        for (int o = 0; o < 10; ++o) acc[o] += __shfl_xor(acc[o], off, 32);
    if (lane == 0) {
        float r = 0.0f;
#pragma unroll
        for (int o = 0; o < 10; ++o) r += (acc[o] + ob[o]) * (float)(1 << o);
        out[wave] = r;
    }
}

// ---------------------------------------------------------------------------
extern "C" void kernel_launch(void* const* d_in, const int* in_sizes, int n_in,
                              void* d_out, int out_size, void* d_ws, size_t ws_size,
                              hipStream_t stream) {
    const float* x        = (const float*)d_in[0];
    const int*   eidx     = (const int*)d_in[1];
    const float* eattr    = (const float*)d_in[2];
    const float* embed_W  = (const float*)d_in[3];
    const float* embed_b  = (const float*)d_in[4];
    const float* edge_W   = (const float*)d_in[5];
    const float* edge_b   = (const float*)d_in[6];
    const float* W1       = (const float*)d_in[7];
    const float* b1       = (const float*)d_in[8];
    const float* W2       = (const float*)d_in[9];
    const float* b2       = (const float*)d_in[10];
    const float* W3       = (const float*)d_in[11];
    const float* b3       = (const float*)d_in[12];
    const float* out_W    = (const float*)d_in[13];
    const float* out_b    = (const float*)d_in[14];
    float* out = (float*)d_out;

    const int N = in_sizes[0] / 16;       // 20000 nodes
    const int E = in_sizes[2] / EDGE_DIM; // 640000 edges
    const int* src = eidx;                // edge_index[0]
    const int* dst = eidx + E;            // edge_index[1]

    char* base = (char*)d_ws;
    size_t off = 0;
    auto alloc = [&](size_t bytes) { void* p = base + off; off = (off + bytes + 255) & ~(size_t)255; return p; };
    float*    ea_log   = (float*)alloc((size_t)E * EDGE_DIM * 4);
    float*    xlog     = (float*)alloc((size_t)N * 16 * 4);
    float*    h        = (float*)alloc((size_t)N * HID * 4);
    float*    x2       = (float*)alloc((size_t)N * HID * 4);
    float*    z        = (float*)alloc((size_t)N * HID * 4);
    float*    t        = (float*)alloc((size_t)N * HID * 4);
    _Float16* Wp       = (_Float16*)alloc((size_t)N_BLOCKS * 3 * HID * HID * 2);
    int*      counts   = (int*)alloc((size_t)N * 4);
    int*      row_start= (int*)alloc((size_t)(N + 1) * 4);
    int*      cursor   = (int*)alloc((size_t)N * 4);
    int*      edge_ids = (int*)alloc((size_t)E * 4);
    (void)ws_size; (void)n_in; (void)out_size;

    (void)hipMemsetAsync(counts, 0, (size_t)N * 4, stream);
    k_log1p<<<(N * 16 + 255) / 256, 256, 0, stream>>>(x, xlog, N * 16);
    k_log1p<<<(E * EDGE_DIM + 255) / 256, 256, 0, stream>>>(eattr, ea_log, E * EDGE_DIM);
    k_embed<<<(N * HID + 255) / 256, 256, 0, stream>>>(xlog, embed_W, embed_b, h, N);

    k_count<<<(E + 255) / 256, 256, 0, stream>>>(dst, counts, E);
    k_scan<<<1, 1024, 0, stream>>>(counts, row_start, cursor, N);
    k_fill<<<(E + 255) / 256, 256, 0, stream>>>(dst, cursor, edge_ids, E);

    const int pack_total = N_BLOCKS * 3 * HID * HID;
    k_pack<<<(pack_total + 255) / 256, 256, 0, stream>>>(W1, W2, W3, Wp, pack_total);

    const int norm_blocks = (N + 7) / 8;             // 8 waves per 256-thread block
    const int gemm_blocks = N / (16 * MT_PER_BLK);   // 625

    for (int b = 0; b < N_BLOCKS; ++b) {
        k_norm<<<norm_blocks, 256, 0, stream>>>(h, x2, N);
        k_aggregate<<<norm_blocks, 256, 0, stream>>>(
            x2, ea_log, src, row_start, edge_ids,
            edge_W + (size_t)b * EDGE_DIM * HID, edge_b + (size_t)b * HID, z, N);
        k_gemm128<<<gemm_blocks, 256, 0, stream>>>(z, Wp + ((size_t)(b * 3 + 0)) * HID * HID,
                                                   b1 + (size_t)b * HID, t, 1);
        k_gemm128<<<gemm_blocks, 256, 0, stream>>>(t, Wp + ((size_t)(b * 3 + 1)) * HID * HID,
                                                   b2 + (size_t)b * HID, z, 1);
        k_gemm128<<<gemm_blocks, 256, 0, stream>>>(z, Wp + ((size_t)(b * 3 + 2)) * HID * HID,
                                                   b3 + (size_t)b * HID, t, 0);
        k_norm_res<<<norm_blocks, 256, 0, stream>>>(t, h, N);
    }

    k_out<<<norm_blocks, 256, 0, stream>>>(h, out_W, out_b, out, N);
}
